// DGCN_41807211659522
// MI455X (gfx1250) — matrix operand order repaired and checked by reading, not compiled, not run on hardware
//
#include <hip/hip_runtime.h>
#include <stdint.h>

// ---------- CDNA5 WMMA types ----------
typedef __attribute__((ext_vector_type(16))) __bf16   v16bf;
typedef __attribute__((ext_vector_type(8)))  float    v8f;
typedef __attribute__((ext_vector_type(4)))  unsigned v4u;
typedef __attribute__((ext_vector_type(4)))  unsigned uint32x4;
typedef __attribute__((ext_vector_type(8)))  int      int32x8;
typedef __attribute__((ext_vector_type(4)))  int      int32x4;

union FragB16 { unsigned u[8]; v16bf v; };

// Problem constants
#define BB   32
#define NN   512
#define TT   12
#define CC   64
#define TC   768      // T*C
#define NT   6144     // N*T
#define CIN  448      // 7*C

#define LDS_STRIDE 40   // 32 bf16 payload + 4-dword pad -> 80B rows (bank-conflict free, 16B aligned)

__device__ __forceinline__ unsigned short f32_bf16(float f) {
    union { float f; unsigned u; } c; c.f = f;
    unsigned u = c.u;
    u += 0x7FFFu + ((u >> 16) & 1u);      // round-to-nearest-even
    return (unsigned short)(u >> 16);
}

// A-fragment (16-bit, 16x32) per-lane packed-pair K for VGPR i, half = lane>>4
__device__ __forceinline__ int kA(int i, int half) {
    return (i < 4 ? 2 * i : 16 + 2 * (i - 4)) + half * 8;
}
// B-fragment (16-bit, 32x16): same K order as A fragment
__device__ __forceinline__ int kB(int i, int half) {
    return (i < 4 ? 2 * i : 16 + 2 * (i - 4)) + half * 8;
}

// ---------------- Tensor Data Mover: DMA a 128(row) x 32(bf16) tile into LDS ----------------
// Tile rows are 64B contiguous in global (row stride 512 bf16); LDS rows padded to 80B
// via D# pad fields (pad_interval: 16 DWORDs, pad_amount: 4 DWORDs).
__device__ __forceinline__ void tdm_load_tile(unsigned lds_addr, const unsigned short* gptr) {
    uint64_t ga = (uint64_t)(uintptr_t)gptr;
    uint32x4 g0;
    g0[0] = 1u;                                                  // count=1, user descriptor
    g0[1] = lds_addr;                                            // LDS byte address
    g0[2] = (unsigned)ga;                                        // global_addr[31:0]
    g0[3] = (unsigned)((ga >> 32) & 0x01FFFFFFu) | (2u << 30);   // global_addr[56:32], type=2
    int32x8 g1;
    g1[0] = (1 << 16)      // data_size = 1 -> 2 bytes
          | (1 << 20)      // pad_enable
          | (3 << 22)      // pad_interval = 3 -> every 16 DWORDs
          | (3 << 25);     // pad_amount   = 3 -> 4 DWORDs
    g1[1] = (int)(512u << 16);   // tensor_dim0 = 512 (bits 79:48, low 16 here)
    g1[2] = 0;                   // tensor_dim0 hi, tensor_dim1 low16 (tensor_dim1 = 65536)
    g1[3] = (32 << 16) | 1;      // tensor_dim1 hi16 = 1, tile_dim0 = 32
    g1[4] = 128;                 // tile_dim1 = 128, tile_dim2 = 0
    g1[5] = 512;                 // tensor_dim0_stride = 512 (elements)
    g1[6] = 0;
    g1[7] = 0;
    int32x4 z4 = {0, 0, 0, 0};             // 2-D tensor: groups 2/3 unused
    int32x8 z8 = {0, 0, 0, 0, 0, 0, 0, 0}; // extra group (clang-23 6-arg form)
    __builtin_amdgcn_tensor_load_to_lds(g0, g1, z4, z4, z8, 0);
}

// ---------------- conversion kernels ----------------
__global__ void conv_x(const float* __restrict__ x,
                       unsigned short* __restrict__ xnat,   // chunk 0: [b][nt][C]
                       unsigned short* __restrict__ xT) {   // [b][tc][v]
    size_t idx = (size_t)blockIdx.x * 256 + threadIdx.x;    // < B*N*TC
    unsigned short h = f32_bf16(x[idx]);
    xnat[idx] = h;
    size_t b   = idx / ((size_t)NN * TC);
    size_t rem = idx - b * ((size_t)NN * TC);
    size_t v   = rem / TC;
    size_t tc  = rem - v * TC;
    xT[(b * TC + tc) * NN + v] = h;
}

__global__ void conv_a(const float* __restrict__ a0, const float* __restrict__ a1,
                       const float* __restrict__ a2, unsigned short* __restrict__ at) {
    size_t idx = (size_t)blockIdx.x * 256 + threadIdx.x;    // < B*N*N
    int s = blockIdx.y;
    const float* a = (s == 0) ? a0 : ((s == 1) ? a1 : a2);
    float f = a[idx];
    size_t b   = idx >> 18;          // /(512*512)
    size_t rem = idx & 262143u;
    size_t v   = rem >> 9;
    size_t w   = rem & 511u;
    // a_t[s][b][w][v] = a[b][v][w]
    at[(size_t)s * ((size_t)BB * NN * NN) + ((b << 9) + w) * NN + v] = f32_bf16(f);
}

__global__ void conv_w(const float* __restrict__ W, unsigned short* __restrict__ Wb) {
    int idx = blockIdx.x * 256 + threadIdx.x;
    if (idx < CC * CIN) Wb[idx] = f32_bf16(W[idx]);
}

// ---------------- diffusion hop: out[w][tc] = sum_v aT[w][v] * src[v][tc] ----------------
// TDM double-buffered LDS pipeline; 8 waves (4 along M x 2 along N), wave tile 32x64.
__global__ __launch_bounds__(256) void dgcn_nconv(
    const unsigned short* __restrict__ At,     // [B][512][512]  aT row-major
    const unsigned short* __restrict__ BT,     // [B][768][512]  srcT
    unsigned short* __restrict__ outNat,       // [B][512][768]
    unsigned short* __restrict__ outT,         // [B][768][512]
    int writeT)
{
    __shared__ unsigned short ldsA[2][128 * LDS_STRIDE];
    __shared__ unsigned short ldsB[2][128 * LDS_STRIDE];

    const int tid  = threadIdx.x;
    const int lane = tid & 31, wave = tid >> 5;
    const int half = lane >> 4, r = lane & 15;
    const int b   = blockIdx.z;
    const int m0w = (wave & 3) * 32;                 // wave M offset within block tile
    const int n0w = (wave >> 2) * 64;                // wave N offset within block tile

    // block-tile row starts (k advances along rows)
    const unsigned short* Ag = At + (size_t)b * (NN * NN) + (size_t)(blockIdx.x * 128) * NN;
    const unsigned short* Bg = BT + (size_t)b * (TC * NN) + (size_t)(blockIdx.y * 128) * NN;

    if (wave == 0) {
        tdm_load_tile((unsigned)(uintptr_t)&ldsA[0][0], Ag);
        tdm_load_tile((unsigned)(uintptr_t)&ldsB[0][0], Bg);
    }

    v8f zero = {0.f, 0.f, 0.f, 0.f, 0.f, 0.f, 0.f, 0.f};
    v8f acc[2][4];
#pragma unroll
    for (int mi = 0; mi < 2; mi++)
#pragma unroll
        for (int ni = 0; ni < 4; ni++) acc[mi][ni] = zero;

    for (int kt = 0; kt < 16; kt++) {
        const int cur = kt & 1;
        if (wave == 0) {
            if (kt + 1 < 16) {       // prefetch next tiles into the other buffer
                tdm_load_tile((unsigned)(uintptr_t)&ldsA[1 - cur][0], Ag + (kt + 1) * 32);
                tdm_load_tile((unsigned)(uintptr_t)&ldsB[1 - cur][0], Bg + (kt + 1) * 32);
                __builtin_amdgcn_s_wait_tensorcnt(2);   // current pair landed, next in flight
            } else {
                __builtin_amdgcn_s_wait_tensorcnt(0);
            }
        }
        __syncthreads();                                 // LDS buf[cur] visible to all waves

        const unsigned short* Al = &ldsA[cur][0];
        const unsigned short* Bl = &ldsB[cur][0];
        FragB16 A[2], Bf[4];
#pragma unroll
        for (int mi = 0; mi < 2; mi++) {
            const unsigned short* rp = Al + (size_t)(m0w + mi * 16 + r) * LDS_STRIDE;
#pragma unroll
            for (int i = 0; i < 8; i++)
                A[mi].u[i] = *(const unsigned*)(rp + kA(i, half));
        }
#pragma unroll
        for (int ni = 0; ni < 4; ni++) {
            const unsigned short* cp = Bl + (size_t)(n0w + ni * 16 + r) * LDS_STRIDE;
#pragma unroll
            for (int i = 0; i < 8; i++)
                Bf[ni].u[i] = *(const unsigned*)(cp + kB(i, half));
        }
#pragma unroll
        for (int mi = 0; mi < 2; mi++)
#pragma unroll
            for (int ni = 0; ni < 4; ni++)
                acc[mi][ni] = __builtin_amdgcn_wmma_f32_16x16x32_bf16(
                    false, A[mi].v, false, Bf[ni].v, (short)0, acc[mi][ni], false, false);

        __syncthreads();                                 // all reads of buf[cur] done before reuse
    }

    const int m0 = blockIdx.x * 128 + m0w;
    const int n0 = blockIdx.y * 128 + n0w;

    // natural-layout store (projection chunk input)
    unsigned short* On = outNat + (size_t)b * (NN * TC);
#pragma unroll
    for (int mi = 0; mi < 2; mi++)
#pragma unroll
        for (int ni = 0; ni < 4; ni++) {
            int n = n0 + ni * 16 + r;
#pragma unroll
            for (int r8 = 0; r8 < 8; r8++) {
                int m = m0 + mi * 16 + half * 8 + r8;
                On[(size_t)m * TC + n] = f32_bf16(acc[mi][ni][r8]);
            }
        }

    // transposed store: outT[tc][w] -- 16B contiguous per lane straight from D fragment
    if (writeT) {
        unsigned short* Ot = outT + (size_t)b * (TC * NN);
#pragma unroll
        for (int mi = 0; mi < 2; mi++)
#pragma unroll
            for (int ni = 0; ni < 4; ni++) {
                int n  = n0 + ni * 16 + r;
                int mb = m0 + mi * 16 + half * 8;
                v4u val;
#pragma unroll
                for (int j = 0; j < 4; j++)
                    val[j] = (unsigned)f32_bf16(acc[mi][ni][2 * j]) |
                             ((unsigned)f32_bf16(acc[mi][ni][2 * j + 1]) << 16);
                *(v4u*)(Ot + (size_t)n * NN + mb) = val;
            }
    }
}

// ---------------- projection: y[nt][o] = sum_c h[nt][c] * W[o][c] + bias[o] ----------------
__global__ __launch_bounds__(256) void dgcn_proj(
    const unsigned short* __restrict__ chunks,  // 7 x [B][6144][64] bf16
    const unsigned short* __restrict__ Wb,      // [64][448] bf16
    const float* __restrict__ bias,             // [64]
    float* __restrict__ y)                      // [B][6144][64]
{
    const int tid  = threadIdx.x;
    const int lane = tid & 31, wave = tid >> 5;
    const int half = lane >> 4, r = lane & 15;
    const int b  = blockIdx.y;
    const int m0 = blockIdx.x * 256 + wave * 32;        // nt tile (8 waves stacked on M)
    const size_t chunkSz = (size_t)BB * NT * CC;        // 12,582,912
    const size_t bOff    = (size_t)b * NT * CC;

    v8f zero = {0.f, 0.f, 0.f, 0.f, 0.f, 0.f, 0.f, 0.f};
    v8f acc[2][4];
#pragma unroll
    for (int mi = 0; mi < 2; mi++)
#pragma unroll
        for (int ni = 0; ni < 4; ni++) acc[mi][ni] = zero;

    for (int kt = 0; kt < 14; kt++) {                   // 14 K-tiles of 32 over CIN=448
        const unsigned short* cb = chunks + (size_t)(kt >> 1) * chunkSz + bOff + (kt & 1) * 32;
        FragB16 A[2], Bf[4];
#pragma unroll
        for (int mi = 0; mi < 2; mi++) {
            const unsigned short* rp = cb + (size_t)(m0 + mi * 16 + r) * CC;
#pragma unroll
            for (int i = 0; i < 8; i++)
                A[mi].u[i] = *(const unsigned*)(rp + kA(i, half));
        }
#pragma unroll
        for (int ni = 0; ni < 4; ni++) {
            const unsigned short* wp = Wb + (size_t)(ni * 16 + r) * CIN + kt * 32;
#pragma unroll
            for (int i = 0; i < 8; i++)
                Bf[ni].u[i] = *(const unsigned*)(wp + kB(i, half));
        }
#pragma unroll
        for (int mi = 0; mi < 2; mi++)
#pragma unroll
            for (int ni = 0; ni < 4; ni++)
                acc[mi][ni] = __builtin_amdgcn_wmma_f32_16x16x32_bf16(
                    false, A[mi].v, false, Bf[ni].v, (short)0, acc[mi][ni], false, false);
    }

    float* Y = y + bOff;
#pragma unroll
    for (int ni = 0; ni < 4; ni++) {
        int o = ni * 16 + r;
        float bv = bias[o];
#pragma unroll
        for (int mi = 0; mi < 2; mi++)
#pragma unroll
            for (int r8 = 0; r8 < 8; r8++) {
                int m = m0 + mi * 16 + half * 8 + r8;
                Y[(size_t)m * CC + o] = acc[mi][ni][r8] + bv;   // coalesced across lanes
            }
    }
}

// ---------------- host launcher ----------------
extern "C" void kernel_launch(void* const* d_in, const int* in_sizes, int n_in,
                              void* d_out, int out_size, void* d_ws, size_t ws_size,
                              hipStream_t stream) {
    (void)in_sizes; (void)n_in; (void)out_size; (void)ws_size;
    const float* x  = (const float*)d_in[0];
    const float* a0 = (const float*)d_in[1];
    const float* a1 = (const float*)d_in[2];
    const float* a2 = (const float*)d_in[3];
    const float* W  = (const float*)d_in[4];
    const float* bv = (const float*)d_in[5];
    float* y = (float*)d_out;

    // workspace layout (u16 elements)
    const size_t atSz    = (size_t)BB * NN * NN;        //  8,388,608 per support
    const size_t chunkSz = (size_t)BB * NT * CC;        // 12,582,912 per chunk
    unsigned short* ws     = (unsigned short*)d_ws;
    unsigned short* at     = ws;                        // 3 * atSz
    unsigned short* chunks = at + 3 * atSz;             // 7 * chunkSz (chunk0 = x bf16)
    unsigned short* xT     = chunks + 7 * chunkSz;      // chunkSz
    unsigned short* hopT   = xT + chunkSz;              // chunkSz
    unsigned short* Wb     = hopT + chunkSz;            // 28,672

    conv_x<<<(unsigned)(chunkSz / 256), 256, 0, stream>>>(x, chunks, xT);
    dim3 ga((unsigned)(atSz / 256), 3);
    conv_a<<<ga, 256, 0, stream>>>(a0, a1, a2, at);
    conv_w<<<(CC * CIN + 255) / 256, 256, 0, stream>>>(W, Wb);

    dim3 g(NN / 128, TC / 128, BB);   // 4 x 6 x 32
    for (int s = 0; s < 3; s++) {
        // hop 1: chunk index 1+2s; also emit transposed buffer for hop 2
        dgcn_nconv<<<g, 256, 0, stream>>>(at + (size_t)s * atSz, xT,
                                          chunks + (size_t)(1 + 2 * s) * chunkSz, hopT, 1);
        // hop 2: chunk index 2+2s
        dgcn_nconv<<<g, 256, 0, stream>>>(at + (size_t)s * atSz, hopT,
                                          chunks + (size_t)(2 + 2 * s) * chunkSz, hopT, 0);
    }

    dim3 gp(NT / 256, BB);            // 24 x 32
    dgcn_proj<<<gp, 256, 0, stream>>>(chunks, Wb, bv, y);
}
